// BiAttention_69389491634466
// MI455X (gfx1250) — matrix operand order repaired and checked by reading, not compiled, not run on hardware
//
#include <hip/hip_runtime.h>

#define BB 16
#define LL 2048
#define DD 1024
#define NEGV (-1e20f)

typedef __attribute__((ext_vector_type(16))) __bf16 v16bf;
typedef __attribute__((ext_vector_type(8)))  float  v8f;

struct alignas(16) U128 { unsigned a, b, c, d; };
struct U256 { U128 lo, hi; };

__device__ __forceinline__ __bf16 f2bf(float x) {
  unsigned u = __builtin_bit_cast(unsigned, x);
  unsigned r = (u + 0x7FFFu + ((u >> 16) & 1u)) >> 16;
  unsigned short h = (unsigned short)r;
  return __builtin_bit_cast(__bf16, h);
}

__device__ __forceinline__ v8f zero8() {
  v8f c = {0.f, 0.f, 0.f, 0.f, 0.f, 0.f, 0.f, 0.f};
  return c;
}

// Load a v16bf fragment from two 16-byte-aligned 8-element segments.
__device__ __forceinline__ v16bf ld_bf16_pair(const __bf16* p0, const __bf16* p1) {
  U256 t;
  t.lo = *(const U128*)p0;
  t.hi = *(const U128*)p1;
  return __builtin_bit_cast(v16bf, t);
}
// Contiguous 32-byte (16 x bf16) fragment.
__device__ __forceinline__ v16bf ld_bf16_32B(const __bf16* p) {
  return ld_bf16_pair(p, p + 8);
}

// ======== fallback (on-the-fly f32->bf16) fragment loaders ==============
__device__ __forceinline__ v16bf load_a_frag(const float* base, int ldr, int k0,
                                             const float* scale, int lane) {
  int row = lane & 15;
  int kh  = lane >> 4;
  const float* p = base + (size_t)row * ldr;
  int kA = k0 + kh * 16;
  int kB = k0 + kh * 16 + 8;
  v16bf a;
#pragma unroll
  for (int j = 0; j < 8; ++j) {
    a[j]     = f2bf(p[kA + j] * scale[kA + j]);
    a[8 + j] = f2bf(p[kB + j] * scale[kB + j]);
  }
  return a;
}
__device__ __forceinline__ v16bf load_bT_frag(const float* mbase, int ldr, int k0, int lane) {
  int col = lane & 15;
  int kh  = lane >> 4;
  const float* p = mbase + (size_t)col * ldr + k0 + kh * 16;
  v16bf b;
#pragma unroll
  for (int j = 0; j < 16; ++j) b[j] = f2bf(p[j]);
  return b;
}
__device__ __forceinline__ v16bf load_b_frag(const float* vbase, int ldr, int lane) {
  int col = lane & 15;
  int kh  = lane >> 4;
  const float* p = vbase + (size_t)(kh * 16) * ldr + col;
  v16bf b;
#pragma unroll
  for (int j = 0; j < 16; ++j) b[j] = f2bf(p[(size_t)j * ldr]);
  return b;
}

// ---------------- Pass 0a: row dot products ------------------------------
__global__ void bi_dot_kernel(const float* __restrict__ X, const float* __restrict__ w,
                              float* __restrict__ out, int nrows) {
  int wid  = (blockIdx.x * blockDim.x + threadIdx.x) >> 5;
  int lane = threadIdx.x & 31;
  if (wid >= nrows) return;
  const float* row = X + (size_t)wid * DD;
  float s = 0.f;
  for (int d = lane; d < DD; d += 32) s += row[d] * w[d];
#pragma unroll
  for (int off = 16; off > 0; off >>= 1) s += __shfl_xor(s, off, 32);
  if (lane == 0) out[wid] = s;
}

// ---------------- Pass 0b: bf16 staging ----------------------------------
__global__ void bi_cvt_scale_kernel(const float* __restrict__ x, const float* __restrict__ scale,
                                    __bf16* __restrict__ y) {
  size_t i0 = ((size_t)blockIdx.x * 256 + threadIdx.x) * 4;
#pragma unroll
  for (int j = 0; j < 4; ++j) {
    size_t i = i0 + j;
    y[i] = f2bf(x[i] * scale[i & (DD - 1)]);
  }
}

__global__ void bi_cvt_kernel(const float* __restrict__ x, __bf16* __restrict__ y) {
  size_t i0 = ((size_t)blockIdx.x * 256 + threadIdx.x) * 4;
#pragma unroll
  for (int j = 0; j < 4; ++j) {
    size_t i = i0 + j;
    y[i] = f2bf(x[i]);
  }
}

// Tb[b][d][m] = bf16(memory[b][m][d]) via LDS tile transpose
__global__ void bi_transpose_kernel(const float* __restrict__ mem, __bf16* __restrict__ Tb) {
  __shared__ __bf16 tile[32][33];
  int b  = blockIdx.z;
  int m0 = blockIdx.y * 32;
  int d0 = blockIdx.x * 32;
  int tx = threadIdx.x;      // 0..31
  int ty = threadIdx.y;      // 0..7
#pragma unroll
  for (int i = 0; i < 32; i += 8) {
    int m = m0 + ty + i;
    tile[ty + i][tx] = f2bf(mem[((size_t)b * LL + m) * DD + d0 + tx]);
  }
  __syncthreads();
#pragma unroll
  for (int i = 0; i < 32; i += 8) {
    int d = d0 + ty + i;
    Tb[((size_t)b * DD + d) * LL + m0 + tx] = tile[tx][ty + i];
  }
}

// ---------------- Pass 1: per-row softmax stats (staged bf16) ------------
__global__ __launch_bounds__(256) void bi_stats_bf16(
    const __bf16* __restrict__ Ab, const __bf16* __restrict__ Mb,
    const unsigned char* __restrict__ mask,
    const float* __restrict__ idot, const float* __restrict__ mdot,
    float* __restrict__ rowmax, float* __restrict__ rowsum) {
  int lane = threadIdx.x & 31;
  int wave = threadIdx.x >> 5;
  int b    = blockIdx.y;
  int lt   = blockIdx.x * 8 + wave;
  int l0   = lt * 16;
  int kh   = lane >> 4;

  const __bf16* Arow = Ab + ((size_t)b * LL + l0 + (lane & 15)) * DD;

  v16bf afr[32];
#pragma unroll
  for (int kc = 0; kc < 32; ++kc)
    afr[kc] = ld_bf16_32B(Arow + kc * 32 + kh * 16);

  float id8[8], mL8[8];
#pragma unroll
  for (int r = 0; r < 8; ++r) {
    int row = l0 + r + 8 * kh;
    id8[r] = idot[b * LL + row];
    mL8[r] = mask[b * LL + row] ? 1.f : 0.f;
  }

  float m_r[8], l_r[8];
#pragma unroll
  for (int r = 0; r < 8; ++r) { m_r[r] = -3e38f; l_r[r] = 0.f; }

  for (int m0 = 0; m0 < LL; m0 += 16) {
    const __bf16* Mrow = Mb + ((size_t)b * LL + m0 + (lane & 15)) * DD;
    v8f c = zero8();
#pragma unroll
    for (int kc = 0; kc < 32; ++kc) {
      v16bf bfr = ld_bf16_32B(Mrow + kc * 32 + kh * 16);
      c = __builtin_amdgcn_wmma_f32_16x16x32_bf16(false, afr[kc], false, bfr,
                                                  (short)0, c, false, false);
    }
    int   col = m0 + (lane & 15);
    float md  = mdot[b * LL + col];
    bool  mM  = mask[b * LL + col] != 0;
#pragma unroll
    for (int r = 0; r < 8; ++r) {
      float v = c[r] + id8[r] + md;
      if (mM || mL8[r] > 0.5f) v = NEGV;
      float mx = v;
      mx = fmaxf(mx, __shfl_xor(mx, 1, 16));
      mx = fmaxf(mx, __shfl_xor(mx, 2, 16));
      mx = fmaxf(mx, __shfl_xor(mx, 4, 16));
      mx = fmaxf(mx, __shfl_xor(mx, 8, 16));
      float newm = fmaxf(m_r[r], mx);
      float p = __expf(v - newm);
      p += __shfl_xor(p, 1, 16);
      p += __shfl_xor(p, 2, 16);
      p += __shfl_xor(p, 4, 16);
      p += __shfl_xor(p, 8, 16);
      l_r[r] = l_r[r] * __expf(m_r[r] - newm) + p;
      m_r[r] = newm;
    }
  }
#pragma unroll
  for (int r = 0; r < 8; ++r) {
    if ((lane & 15) == r) {
      int row = l0 + r + 8 * kh;
      rowmax[b * LL + row] = m_r[r];
      rowsum[b * LL + row] = l_r[r];
    }
  }
}

// ---------------- Pass 2: output_one + first 3 slices (staged) -----------
__global__ __launch_bounds__(256) void bi_out_bf16(
    const __bf16* __restrict__ Ab, const __bf16* __restrict__ Mb,
    const __bf16* __restrict__ Tb, const float* __restrict__ input,
    const unsigned char* __restrict__ mask,
    const float* __restrict__ idot, const float* __restrict__ mdot,
    const float* __restrict__ rowmax, const float* __restrict__ rowsum,
    float* __restrict__ out) {
  __shared__ __bf16 Pl[16][136];

  int lane = threadIdx.x & 31;
  int wave = threadIdx.x >> 5;
  int b    = blockIdx.y;
  int l0   = blockIdx.x * 16;
  int kh   = lane >> 4;

  const __bf16* Arow = Ab + ((size_t)b * LL + l0 + (lane & 15)) * DD;

  v16bf afr[32];
#pragma unroll
  for (int kc = 0; kc < 32; ++kc)
    afr[kc] = ld_bf16_32B(Arow + kc * 32 + kh * 16);

  float im8[8], is8[8], id8[8], mL8[8];
#pragma unroll
  for (int r = 0; r < 8; ++r) {
    int row = l0 + r + 8 * kh;
    im8[r] = rowmax[b * LL + row];
    is8[r] = 1.f / rowsum[b * LL + row];
    id8[r] = idot[b * LL + row];
    mL8[r] = mask[b * LL + row] ? 1.f : 0.f;
  }

  v8f acc[8];
#pragma unroll
  for (int t = 0; t < 8; ++t) acc[t] = zero8();

  for (int mb = 0; mb < 16; ++mb) {
    int m0 = mb * 128 + wave * 16;
    const __bf16* Mrow = Mb + ((size_t)b * LL + m0 + (lane & 15)) * DD;
    v8f c = zero8();
#pragma unroll
    for (int kc = 0; kc < 32; ++kc) {
      v16bf bfr = ld_bf16_32B(Mrow + kc * 32 + kh * 16);
      c = __builtin_amdgcn_wmma_f32_16x16x32_bf16(false, afr[kc], false, bfr,
                                                  (short)0, c, false, false);
    }
    int   col = m0 + (lane & 15);
    float md  = mdot[b * LL + col];
    bool  mM  = mask[b * LL + col] != 0;

    __syncthreads();
#pragma unroll
    for (int r = 0; r < 8; ++r) {
      float v = c[r] + id8[r] + md;
      if (mM || mL8[r] > 0.5f) v = NEGV;
      float p = __expf(v - im8[r]) * is8[r];
      Pl[r + 8 * kh][wave * 16 + (lane & 15)] = f2bf(p);
    }
    __syncthreads();

    int d0w = wave * 128;
#pragma unroll
    for (int kc = 0; kc < 4; ++kc) {
      int prow = lane & 15;
      v16bf pa = ld_bf16_32B(&Pl[prow][kc * 32 + kh * 16]);
#pragma unroll
      for (int t = 0; t < 8; ++t) {
        const __bf16* Tcol = Tb + ((size_t)b * DD + d0w + t * 16 + (lane & 15)) * LL;
        v16bf vb = ld_bf16_32B(Tcol + mb * 128 + kc * 32 + kh * 16);
        acc[t] = __builtin_amdgcn_wmma_f32_16x16x32_bf16(false, pa, false, vb,
                                                         (short)0, acc[t], false, false);
      }
    }
  }

  size_t obase = ((size_t)b * LL + l0) * (size_t)(4 * DD);
#pragma unroll
  for (int t = 0; t < 8; ++t) {
    int colg = wave * 128 + t * 16 + (lane & 15);
#pragma unroll
    for (int r = 0; r < 8; ++r) {
      int row = r + 8 * kh;
      float o  = acc[t][r];
      float iv = input[((size_t)b * LL + l0 + row) * DD + colg];
      float* orow = out + obase + (size_t)row * (4 * DD);
      orow[colg]          = iv;
      orow[DD + colg]     = o;
      orow[2 * DD + colg] = iv * o;
    }
  }
}

// ======== fallback hot kernels (round-1 style, no staging) ===============
__global__ __launch_bounds__(256) void bi_stats_f32(
    const float* __restrict__ input, const float* __restrict__ memory,
    const unsigned char* __restrict__ mask, const float* __restrict__ dot_scale,
    const float* __restrict__ idot, const float* __restrict__ mdot,
    float* __restrict__ rowmax, float* __restrict__ rowsum) {
  int lane = threadIdx.x & 31;
  int wave = threadIdx.x >> 5;
  int b    = blockIdx.y;
  int lt   = blockIdx.x * 8 + wave;
  int l0   = lt * 16;
  int kh   = lane >> 4;
  const float* Abase = input + ((size_t)b * LL + l0) * DD;
  v16bf afr[32];
#pragma unroll
  for (int kc = 0; kc < 32; ++kc) afr[kc] = load_a_frag(Abase, DD, kc * 32, dot_scale, lane);
  float id8[8], mL8[8];
#pragma unroll
  for (int r = 0; r < 8; ++r) {
    int row = l0 + r + 8 * kh;
    id8[r] = idot[b * LL + row];
    mL8[r] = mask[b * LL + row] ? 1.f : 0.f;
  }
  float m_r[8], l_r[8];
#pragma unroll
  for (int r = 0; r < 8; ++r) { m_r[r] = -3e38f; l_r[r] = 0.f; }
  for (int m0 = 0; m0 < LL; m0 += 16) {
    const float* Mbase = memory + ((size_t)b * LL + m0) * DD;
    v8f c = zero8();
#pragma unroll
    for (int kc = 0; kc < 32; ++kc) {
      v16bf bfr = load_bT_frag(Mbase, DD, kc * 32, lane);
      c = __builtin_amdgcn_wmma_f32_16x16x32_bf16(false, afr[kc], false, bfr,
                                                  (short)0, c, false, false);
    }
    int   col = m0 + (lane & 15);
    float md  = mdot[b * LL + col];
    bool  mM  = mask[b * LL + col] != 0;
#pragma unroll
    for (int r = 0; r < 8; ++r) {
      float v = c[r] + id8[r] + md;
      if (mM || mL8[r] > 0.5f) v = NEGV;
      float mx = v;
      mx = fmaxf(mx, __shfl_xor(mx, 1, 16));
      mx = fmaxf(mx, __shfl_xor(mx, 2, 16));
      mx = fmaxf(mx, __shfl_xor(mx, 4, 16));
      mx = fmaxf(mx, __shfl_xor(mx, 8, 16));
      float newm = fmaxf(m_r[r], mx);
      float p = __expf(v - newm);
      p += __shfl_xor(p, 1, 16);
      p += __shfl_xor(p, 2, 16);
      p += __shfl_xor(p, 4, 16);
      p += __shfl_xor(p, 8, 16);
      l_r[r] = l_r[r] * __expf(m_r[r] - newm) + p;
      m_r[r] = newm;
    }
  }
#pragma unroll
  for (int r = 0; r < 8; ++r) {
    if ((lane & 15) == r) {
      int row = l0 + r + 8 * kh;
      rowmax[b * LL + row] = m_r[r];
      rowsum[b * LL + row] = l_r[r];
    }
  }
}

__global__ __launch_bounds__(256) void bi_out_f32(
    const float* __restrict__ input, const float* __restrict__ memory,
    const unsigned char* __restrict__ mask, const float* __restrict__ dot_scale,
    const float* __restrict__ idot, const float* __restrict__ mdot,
    const float* __restrict__ rowmax, const float* __restrict__ rowsum,
    float* __restrict__ out) {
  __shared__ __bf16 Pl[16][136];
  int lane = threadIdx.x & 31;
  int wave = threadIdx.x >> 5;
  int b    = blockIdx.y;
  int l0   = blockIdx.x * 16;
  int kh   = lane >> 4;
  const float* Abase = input + ((size_t)b * LL + l0) * DD;
  float im8[8], is8[8], id8[8], mL8[8];
#pragma unroll
  for (int r = 0; r < 8; ++r) {
    int row = l0 + r + 8 * kh;
    im8[r] = rowmax[b * LL + row];
    is8[r] = 1.f / rowsum[b * LL + row];
    id8[r] = idot[b * LL + row];
    mL8[r] = mask[b * LL + row] ? 1.f : 0.f;
  }
  v8f acc[8];
#pragma unroll
  for (int t = 0; t < 8; ++t) acc[t] = zero8();
  for (int mb = 0; mb < 16; ++mb) {
    int m0 = mb * 128 + wave * 16;
    const float* Mbase = memory + ((size_t)b * LL + m0) * DD;
    v8f c = zero8();
#pragma unroll
    for (int kc = 0; kc < 32; ++kc) {
      v16bf afr = load_a_frag(Abase, DD, kc * 32, dot_scale, lane);
      v16bf bfr = load_bT_frag(Mbase, DD, kc * 32, lane);
      c = __builtin_amdgcn_wmma_f32_16x16x32_bf16(false, afr, false, bfr,
                                                  (short)0, c, false, false);
    }
    int   col = m0 + (lane & 15);
    float md  = mdot[b * LL + col];
    bool  mM  = mask[b * LL + col] != 0;
    __syncthreads();
#pragma unroll
    for (int r = 0; r < 8; ++r) {
      float v = c[r] + id8[r] + md;
      if (mM || mL8[r] > 0.5f) v = NEGV;
      float p = __expf(v - im8[r]) * is8[r];
      Pl[r + 8 * kh][wave * 16 + (lane & 15)] = f2bf(p);
    }
    __syncthreads();
    int d0w = wave * 128;
#pragma unroll
    for (int kc = 0; kc < 4; ++kc) {
      int prow = lane & 15;
      v16bf pa = ld_bf16_32B(&Pl[prow][kc * 32 + kh * 16]);
      const float* Vb = memory + ((size_t)b * LL + mb * 128 + kc * 32) * DD + d0w;
#pragma unroll
      for (int t = 0; t < 8; ++t) {
        v16bf vb = load_b_frag(Vb + t * 16, DD, lane);
        acc[t] = __builtin_amdgcn_wmma_f32_16x16x32_bf16(false, pa, false, vb,
                                                         (short)0, acc[t], false, false);
      }
    }
  }
  size_t obase = ((size_t)b * LL + l0) * (size_t)(4 * DD);
#pragma unroll
  for (int t = 0; t < 8; ++t) {
    int colg = wave * 128 + t * 16 + (lane & 15);
#pragma unroll
    for (int r = 0; r < 8; ++r) {
      int row = r + 8 * kh;
      float o  = acc[t][r];
      float iv = input[((size_t)b * LL + l0 + row) * DD + colg];
      float* orow = out + obase + (size_t)row * (4 * DD);
      orow[colg]          = iv;
      orow[DD + colg]     = o;
      orow[2 * DD + colg] = iv * o;
    }
  }
}

// ---------------- Pass 3: weight_two softmax + output_two ----------------
__global__ __launch_bounds__(256) void bi_w2_kernel(const float* __restrict__ input,
                                                    const float* __restrict__ rowmax,
                                                    float* __restrict__ otwo) {
  __shared__ float w2[LL];
  __shared__ float red[256];
  int b = blockIdx.x;
  int t = threadIdx.x;

  float mx = -3e38f;
  for (int l = t; l < LL; l += 256) mx = fmaxf(mx, rowmax[b * LL + l]);
  red[t] = mx;
  __syncthreads();
  for (int s = 128; s > 0; s >>= 1) {
    if (t < s) red[t] = fmaxf(red[t], red[t + s]);
    __syncthreads();
  }
  float gmax = red[0];
  __syncthreads();

  float sm = 0.f;
  for (int l = t; l < LL; l += 256) {
    float e = __expf(rowmax[b * LL + l] - gmax);
    w2[l] = e;
    sm += e;
  }
  red[t] = sm;
  __syncthreads();
  for (int s = 128; s > 0; s >>= 1) {
    if (t < s) red[t] += red[t + s];
    __syncthreads();
  }
  float inv = 1.f / red[0];

  float acc0 = 0.f, acc1 = 0.f, acc2 = 0.f, acc3 = 0.f;
  for (int l = 0; l < LL; ++l) {
    float w = w2[l] * inv;
    const float* row = input + ((size_t)b * LL + l) * DD;
    acc0 += w * row[t];
    acc1 += w * row[t + 256];
    acc2 += w * row[t + 512];
    acc3 += w * row[t + 768];
  }
  otwo[b * DD + t]       = acc0;
  otwo[b * DD + t + 256] = acc1;
  otwo[b * DD + t + 512] = acc2;
  otwo[b * DD + t + 768] = acc3;
}

// ---------------- Pass 4: out[3D:4D] = output_two * output_one -----------
__global__ void bi_mul_kernel(const float* __restrict__ otwo, float* __restrict__ out) {
  size_t bl = blockIdx.x;
  int b = (int)(bl / LL);
  float* orow = out + bl * (size_t)(4 * DD);
  int t = threadIdx.x;
#pragma unroll
  for (int j = 0; j < 4; ++j) {
    int d = t + j * 256;
    orow[3 * DD + d] = otwo[b * DD + d] * orow[DD + d];
  }
}

extern "C" void kernel_launch(void* const* d_in, const int* in_sizes, int n_in,
                              void* d_out, int out_size, void* d_ws, size_t ws_size,
                              hipStream_t stream) {
  const float*         input     = (const float*)d_in[0];
  const float*         memory    = (const float*)d_in[1];
  const unsigned char* mask      = (const unsigned char*)d_in[2];
  const float*         w_input   = (const float*)d_in[3];
  const float*         w_memory  = (const float*)d_in[4];
  const float*         dot_scale = (const float*)d_in[5];
  float* out = (float*)d_out;
  float* ws  = (float*)d_ws;

  const size_t nBL  = (size_t)BB * LL;
  const size_t nBLD = (size_t)BB * LL * DD;
  const size_t statsFloats = 4 * nBL + (size_t)BB * DD;

  float* idot = ws;
  float* mdot = ws + nBL;
  float* rmax = ws + 2 * nBL;
  float* rsum = ws + 3 * nBL;
  float* otwo = ws + 4 * nBL;

  __bf16* Ab = (__bf16*)(ws + statsFloats);   // statsFloats*4 bytes, 16B aligned
  __bf16* Mb = Ab + nBLD;
  __bf16* Tb = Mb + nBLD;

  const size_t need = statsFloats * sizeof(float) + 3 * nBLD * sizeof(__bf16);
  const bool staged = (ws_size >= need);

  int nrows = BB * LL;
  int dotBlocks = (nrows * 32 + 255) / 256;
  bi_dot_kernel<<<dotBlocks, 256, 0, stream>>>(input, w_input, idot, nrows);
  bi_dot_kernel<<<dotBlocks, 256, 0, stream>>>(memory, w_memory, mdot, nrows);

  if (staged) {
    int cvtBlocks = (int)(nBLD / (256 * 4));
    bi_cvt_scale_kernel<<<cvtBlocks, 256, 0, stream>>>(input, dot_scale, Ab);
    bi_cvt_kernel<<<cvtBlocks, 256, 0, stream>>>(memory, Mb);
    bi_transpose_kernel<<<dim3(DD / 32, LL / 32, BB), dim3(32, 8), 0, stream>>>(memory, Tb);

    bi_stats_bf16<<<dim3(16, BB), 256, 0, stream>>>(Ab, Mb, mask, idot, mdot, rmax, rsum);
    bi_out_bf16<<<dim3(LL / 16, BB), 256, 0, stream>>>(Ab, Mb, Tb, input, mask,
                                                       idot, mdot, rmax, rsum, out);
  } else {
    bi_stats_f32<<<dim3(16, BB), 256, 0, stream>>>(input, memory, mask, dot_scale,
                                                   idot, mdot, rmax, rsum);
    bi_out_f32<<<dim3(LL / 16, BB), 256, 0, stream>>>(input, memory, mask, dot_scale,
                                                      idot, mdot, rmax, rsum, out);
  }

  bi_w2_kernel<<<BB, 256, 0, stream>>>(input, rmax, otwo);
  bi_mul_kernel<<<BB * LL, 256, 0, stream>>>(otwo, out);
}
